// MambaLayer_69544110457393
// MI455X (gfx1250) — compile-verified, matrix operand order checked by reading, not run on hardware
//
#include <hip/hip_runtime.h>
#include <hip/hip_bf16.h>

// ---------------- model dims ----------------
#define D_MODEL 1024
#define D_STATE 16
#define D_CONV  4
#define D_INNER 2048
#define DT_RANK 64
#define B_SZ    2
#define SEQ     2048
#define NROWS   (B_SZ * SEQ)   // 4096

typedef __attribute__((ext_vector_type(16))) __bf16 v16bf;
typedef __attribute__((ext_vector_type(8)))  __bf16 v8bf;
typedef __attribute__((ext_vector_type(8)))  float  v8f;

// =====================================================================
// gfx1250 async global->LDS copy (ASYNCcnt-tracked, bypasses VGPRs)
// =====================================================================
__device__ __forceinline__ void async_copy_b128(const void* g, void* l) {
  asm volatile("global_load_async_to_lds_b128 %0, %1, off"
               :
               : "v"((unsigned)(uintptr_t)l), "v"(g)
               : "memory");
}
__device__ __forceinline__ void wait_async_le4() {
  asm volatile("s_wait_asynccnt 0x4" ::: "memory");
}
__device__ __forceinline__ void wait_async_0() {
  asm volatile("s_wait_asynccnt 0x0" ::: "memory");
}

// =====================================================================
// fp32 -> bf16 elementwise convert
// =====================================================================
__global__ __launch_bounds__(256) void cvt_f32_bf16(const float* __restrict__ s,
                                                    __bf16* __restrict__ d, int n) {
  int i = blockIdx.x * 256 + threadIdx.x;
  if (i < n) d[i] = (__bf16)s[i];
}

// =====================================================================
// LayerNorm over last dim (1024), output bf16. One block per row.
// =====================================================================
__global__ __launch_bounds__(256) void ln_kernel(const float* __restrict__ x,
                                                 const float* __restrict__ w,
                                                 const float* __restrict__ b,
                                                 __bf16* __restrict__ out) {
  int row = blockIdx.x;
  int tid = threadIdx.x;
  const float* xr = x + (size_t)row * D_MODEL;
  float v[4];
  float s = 0.f, s2 = 0.f;
#pragma unroll
  for (int i = 0; i < 4; i++) {
    v[i] = xr[tid + i * 256];
    s += v[i];
    s2 += v[i] * v[i];
  }
  __shared__ float ss[256], ss2[256];
  ss[tid] = s; ss2[tid] = s2;
  __syncthreads();
  for (int off = 128; off; off >>= 1) {
    if (tid < off) { ss[tid] += ss[tid + off]; ss2[tid] += ss2[tid + off]; }
    __syncthreads();
  }
  float mu   = ss[0] * (1.0f / D_MODEL);
  float var  = ss2[0] * (1.0f / D_MODEL) - mu * mu;
  float rstd = rsqrtf(var + 1e-5f);
  __bf16* o = out + (size_t)row * D_MODEL;
#pragma unroll
  for (int i = 0; i < 4; i++) {
    int c = tid + i * 256;
    o[c] = (__bf16)((v[i] - mu) * rstd * w[c] + b[c]);
  }
}

// =====================================================================
// WMMA bf16 GEMM:  C[M,N] = A[M,K] * W[N,K]^T  (both row-major)
// Block tile 128x128, KT=32, double-buffered LDS filled by async
// global->LDS copies. 8 waves (2 in M x 4 in N), each wave computes
// 4x2 tiles of 16x16 via v_wmma_f32_16x16x32_bf16.
// MODE 0: C = acc
// MODE 1: C = acc + res[m,n]           (aux = residual, ldaux = stride)
// MODE 2: C = softplus(acc + aux[n])   (per-output-feature bias)
// =====================================================================
#define LDSS 40   // padded LDS row stride (bf16 elems), 16B-aligned, conflict-free

template <int MODE>
__global__ __launch_bounds__(256)
void gemm_bf16_wmma(const __bf16* __restrict__ A, int lda,
                    const __bf16* __restrict__ W, int ldb,
                    float* __restrict__ C, int ldc,
                    const float* __restrict__ aux, int ldaux,
                    int M, int N, int K) {
  __shared__ __bf16 sA[2][128 * LDSS];
  __shared__ __bf16 sB[2][128 * LDSS];

  const int tid   = threadIdx.x;
  const int bm0   = blockIdx.y * 128;
  const int bn0   = blockIdx.x * 128;
  const int wid   = tid >> 5;
  const int lane  = tid & 31;
  const int wm    = wid & 1;    // 0..1  (M direction, 64 rows each)
  const int wn    = wid >> 1;   // 0..3  (N direction, 32 cols each)
  const int lrow  = lane & 15;
  const int lhalf = lane >> 4;

  v8f acc[4][2];
#pragma unroll
  for (int mi = 0; mi < 4; mi++)
#pragma unroll
    for (int ni = 0; ni < 2; ni++)
      acc[mi][ni] = (v8f){0.f,0.f,0.f,0.f,0.f,0.f,0.f,0.f};

  // per-thread copy coordinates: 512 chunks of 8 bf16 per matrix, 2 per thread
  // chunk c: row r = c>>2, k-offset k8 = (c&3)*8
  const int r0  = tid >> 1;               // rows for i=0 chunk pair layout
  (void)r0;

  // issue async copies of one 128x32 K-stage into buffer `buf`
  auto issue_stage = [&](int buf, int k0) {
#pragma unroll
    for (int i = 0; i < 2; i++) {
      int c  = tid + i * 256;
      int r  = c >> 2;
      int k8 = (c & 3) << 3;
      const __bf16* ga = A + (size_t)(bm0 + r) * lda + (k0 + k8);
      async_copy_b128(ga, &sA[buf][r * LDSS + k8]);
      int nr = bn0 + r;
      if (nr > N - 1) nr = N - 1;   // clamp: junk cols are discarded by store guard
      const __bf16* gw = W + (size_t)nr * ldb + (k0 + k8);
      async_copy_b128(gw, &sB[buf][r * LDSS + k8]);
    }
  };

  const int nstage = K >> 5;
  issue_stage(0, 0);

  for (int s = 0; s < nstage; s++) {
    const int buf = s & 1;
    if (s + 1 < nstage) {
      issue_stage(buf ^ 1, (s + 1) << 5);
      wait_async_le4();   // 4 outstanding belong to next stage; current done
    } else {
      wait_async_0();
    }
    __syncthreads();

    // fragments per ISA 7.12.2 16-bit A/B layout:
    // lane: row = lane&15, half = lane>>4;
    // elems 0..7  -> K = half*8 + 0..7       (16B contiguous)
    // elems 8..15 -> K = 16 + half*8 + 0..7  (16B contiguous)
    v16bf afr[4], bfr[2];
#pragma unroll
    for (int mi = 0; mi < 4; mi++) {
      const __bf16* p = &sA[buf][(wm * 64 + mi * 16 + lrow) * LDSS + lhalf * 8];
      v8bf lo = *(const v8bf*)p;
      v8bf hi = *(const v8bf*)(p + 16);
#pragma unroll
      for (int e = 0; e < 8; e++) { afr[mi][e] = lo[e]; afr[mi][e + 8] = hi[e]; }
    }
#pragma unroll
    for (int ni = 0; ni < 2; ni++) {
      const __bf16* p = &sB[buf][(wn * 32 + ni * 16 + lrow) * LDSS + lhalf * 8];
      v8bf lo = *(const v8bf*)p;
      v8bf hi = *(const v8bf*)(p + 16);
#pragma unroll
      for (int e = 0; e < 8; e++) { bfr[ni][e] = lo[e]; bfr[ni][e + 8] = hi[e]; }
    }

#pragma unroll
    for (int mi = 0; mi < 4; mi++)
#pragma unroll
      for (int ni = 0; ni < 2; ni++)
        acc[mi][ni] = __builtin_amdgcn_wmma_f32_16x16x32_bf16(
            false, afr[mi], false, bfr[ni], (short)0, acc[mi][ni], false, false);

    __syncthreads();   // guard LDS reuse before next issue overwrites this buf
  }

  // epilogue: D layout — lane n = lane&15; VGPR r -> m = r + 8*(lane>>4)
#pragma unroll
  for (int mi = 0; mi < 4; mi++) {
#pragma unroll
    for (int ni = 0; ni < 2; ni++) {
      int n = bn0 + wn * 32 + ni * 16 + lrow;
      if (n < N) {
#pragma unroll
        for (int r = 0; r < 8; r++) {
          int m = bm0 + wm * 64 + mi * 16 + r + 8 * lhalf;
          float v = acc[mi][ni][r];
          if (MODE == 1) v += aux[(size_t)m * ldaux + n];
          if (MODE == 2) {
            v += aux[n];
            v = (v > 20.f) ? v : log1pf(__expf(v));
          }
          C[(size_t)m * ldc + n] = v;
        }
      }
    }
  }
}

// =====================================================================
// causal depthwise conv (k=4) + SiLU on x_in (left half of xz)
// xz: [NROWS, 2*D_INNER]; outputs xc fp32 and bf16, [NROWS, D_INNER]
// =====================================================================
__global__ __launch_bounds__(256) void conv_silu_kernel(const float* __restrict__ xz,
                                                        const float* __restrict__ cw,
                                                        const float* __restrict__ cb,
                                                        float* __restrict__ xc,
                                                        __bf16* __restrict__ xcb) {
  int gid = blockIdx.x * 256 + threadIdx.x;   // NROWS*D_INNER
  int d   = gid & (D_INNER - 1);
  int row = gid >> 11;                        // D_INNER = 2048
  int b   = row >> 11;                        // SEQ = 2048
  int t   = row & (SEQ - 1);
  float acc = cb[d];
#pragma unroll
  for (int j = 0; j < D_CONV; j++) {
    int tj = t - (D_CONV - 1) + j;
    if (tj >= 0)
      acc += cw[d * D_CONV + j] * xz[((size_t)(b * SEQ + tj)) * (2 * D_INNER) + d];
  }
  float s = acc / (1.f + __expf(-acc));
  xc[gid]  = s;
  xcb[gid] = (__bf16)s;
}

// =====================================================================
// extract dt columns (0..63) of dbl [NROWS,96] into bf16 [NROWS,64]
// =====================================================================
__global__ __launch_bounds__(256) void dt_extract_kernel(const float* __restrict__ dbl,
                                                         __bf16* __restrict__ dtb) {
  int gid = blockIdx.x * 256 + threadIdx.x;   // NROWS*64
  int c   = gid & 63;
  int row = gid >> 6;
  dtb[gid] = (__bf16)dbl[(size_t)row * (DT_RANK + 2 * D_STATE) + c];
}

// =====================================================================
// selective scan. One thread per (b,d) channel, 16 states in VGPRs.
// B/C (cols 64..95 of dbl) staged through LDS in 64-step chunks.
// dy holds delta on input; overwritten in place with y.
// =====================================================================
#define TCH 64
__global__ __launch_bounds__(256) void scan_kernel(const float* __restrict__ dbl,
                                                   float* __restrict__ dy,
                                                   const float* __restrict__ xc,
                                                   const float* __restrict__ A_log) {
  int b = blockIdx.x >> 3;                        // 8 blocks per batch
  int d = ((blockIdx.x & 7) << 8) + threadIdx.x;  // 0..2047
  float A[D_STATE], h[D_STATE];
#pragma unroll
  for (int n = 0; n < D_STATE; n++) {
    A[n] = -__expf(A_log[(size_t)d * D_STATE + n]);
    h[n] = 0.f;
  }
  __shared__ float sBC[TCH][2 * D_STATE];
  const size_t rb = (size_t)b * SEQ;
  for (int t0 = 0; t0 < SEQ; t0 += TCH) {
    __syncthreads();
    for (int i = threadIdx.x; i < TCH * 2 * D_STATE; i += 256) {
      int r = i >> 5, c = i & 31;
      sBC[r][c] = dbl[(rb + t0 + r) * (DT_RANK + 2 * D_STATE) + DT_RANK + c];
    }
    __syncthreads();
    for (int tt = 0; tt < TCH; tt++) {
      size_t row = rb + t0 + tt;
      float dlt = dy[row * D_INNER + d];
      float u   = xc[row * D_INNER + d];
      float du  = dlt * u;
      float y   = 0.f;
#pragma unroll
      for (int n = 0; n < D_STATE; n++) {
        float dA = __expf(dlt * A[n]);
        h[n] = dA * h[n] + du * sBC[tt][n];
        y += h[n] * sBC[tt][D_STATE + n];
      }
      dy[row * D_INNER + d] = y;
    }
  }
}

// =====================================================================
// gating: yb = bf16( (y + xc*D) * silu(z) )
// =====================================================================
__global__ __launch_bounds__(256) void gate_kernel(const float* __restrict__ y,
                                                   const float* __restrict__ xc,
                                                   const float* __restrict__ xz,
                                                   const float* __restrict__ Dp,
                                                   __bf16* __restrict__ yb) {
  int gid = blockIdx.x * 256 + threadIdx.x;    // NROWS*D_INNER
  int d   = gid & (D_INNER - 1);
  int row = gid >> 11;
  float z = xz[(size_t)row * (2 * D_INNER) + D_INNER + d];
  float g = z / (1.f + __expf(-z));
  yb[gid] = (__bf16)((y[gid] + xc[gid] * Dp[d]) * g);
}

// =====================================================================
// host launch
// =====================================================================
extern "C" void kernel_launch(void* const* d_in, const int* in_sizes, int n_in,
                              void* d_out, int out_size, void* d_ws, size_t ws_size,
                              hipStream_t stream) {
  (void)in_sizes; (void)n_in; (void)out_size; (void)ws_size;
  const float* x         = (const float*)d_in[0];
  const float* ln_w      = (const float*)d_in[1];
  const float* ln_b      = (const float*)d_in[2];
  const float* in_proj_w = (const float*)d_in[3];   // [4096,1024]
  const float* conv_w    = (const float*)d_in[4];   // [2048,1,4]
  const float* conv_b    = (const float*)d_in[5];
  const float* x_proj_w  = (const float*)d_in[6];   // [96,2048]
  const float* dt_proj_w = (const float*)d_in[7];   // [2048,64]
  const float* dt_proj_b = (const float*)d_in[8];
  const float* A_log     = (const float*)d_in[9];   // [2048,16]
  const float* D_param   = (const float*)d_in[10];
  const float* out_proj_w= (const float*)d_in[11];  // [1024,2048]
  float* out = (float*)d_out;

  char* wp = (char*)d_ws;
  auto alloc = [&](size_t bytes) -> void* {
    void* p = (void*)wp;
    wp += (bytes + 255) & ~(size_t)255;
    return p;
  };
  __bf16* winb  = (__bf16*)alloc((size_t)2 * D_INNER * D_MODEL * 2);
  __bf16* woutb = (__bf16*)alloc((size_t)D_MODEL * D_INNER * 2);
  __bf16* wxpb  = (__bf16*)alloc((size_t)(DT_RANK + 2 * D_STATE) * D_INNER * 2);
  __bf16* wdtb  = (__bf16*)alloc((size_t)D_INNER * DT_RANK * 2);
  __bf16* xnb   = (__bf16*)alloc((size_t)NROWS * D_MODEL * 2);
  float*  xz    = (float*) alloc((size_t)NROWS * 2 * D_INNER * 4);
  float*  xc    = (float*) alloc((size_t)NROWS * D_INNER * 4);
  __bf16* xcb   = (__bf16*)alloc((size_t)NROWS * D_INNER * 2);
  float*  dbl   = (float*) alloc((size_t)NROWS * (DT_RANK + 2 * D_STATE) * 4);
  __bf16* dtb   = (__bf16*)alloc((size_t)NROWS * DT_RANK * 2);
  float*  dlt   = (float*) alloc((size_t)NROWS * D_INNER * 4);   // delta, then y
  __bf16* yb    = (__bf16*)alloc((size_t)NROWS * D_INNER * 2);

  // 1. weight conversions
  {
    int n1 = 2 * D_INNER * D_MODEL;
    cvt_f32_bf16<<<(n1 + 255) / 256, 256, 0, stream>>>(in_proj_w, winb, n1);
    int n2 = D_MODEL * D_INNER;
    cvt_f32_bf16<<<(n2 + 255) / 256, 256, 0, stream>>>(out_proj_w, woutb, n2);
    int n3 = (DT_RANK + 2 * D_STATE) * D_INNER;
    cvt_f32_bf16<<<(n3 + 255) / 256, 256, 0, stream>>>(x_proj_w, wxpb, n3);
    int n4 = D_INNER * DT_RANK;
    cvt_f32_bf16<<<(n4 + 255) / 256, 256, 0, stream>>>(dt_proj_w, wdtb, n4);
  }

  // 2. layernorm -> bf16
  ln_kernel<<<NROWS, 256, 0, stream>>>(x, ln_w, ln_b, xnb);

  // 3. in_proj: xz[4096,4096] = xnb[4096,1024] * winb[4096,1024]^T
  gemm_bf16_wmma<0><<<dim3((2 * D_INNER) / 128, NROWS / 128), 256, 0, stream>>>(
      xnb, D_MODEL, winb, D_MODEL, xz, 2 * D_INNER, nullptr, 0,
      NROWS, 2 * D_INNER, D_MODEL);

  // 4. depthwise conv + SiLU
  conv_silu_kernel<<<(NROWS * D_INNER) / 256, 256, 0, stream>>>(xz, conv_w, conv_b, xc, xcb);

  // 5. x_proj: dbl[4096,96] = xcb[4096,2048] * wxpb[96,2048]^T
  gemm_bf16_wmma<0><<<dim3(1, NROWS / 128), 256, 0, stream>>>(
      xcb, D_INNER, wxpb, D_INNER, dbl, DT_RANK + 2 * D_STATE, nullptr, 0,
      NROWS, DT_RANK + 2 * D_STATE, D_INNER);

  // 6. extract dt as bf16
  dt_extract_kernel<<<(NROWS * DT_RANK) / 256, 256, 0, stream>>>(dbl, dtb);

  // 7. dt_proj + bias + softplus: dlt[4096,2048]
  gemm_bf16_wmma<2><<<dim3(D_INNER / 128, NROWS / 128), 256, 0, stream>>>(
      dtb, DT_RANK, wdtb, DT_RANK, dlt, D_INNER, dt_proj_b, 0,
      NROWS, D_INNER, DT_RANK);

  // 8. selective scan (in-place delta -> y)
  scan_kernel<<<B_SZ * (D_INNER / 256), 256, 0, stream>>>(dbl, dlt, xc, A_log);

  // 9. gating -> bf16
  gate_kernel<<<(NROWS * D_INNER) / 256, 256, 0, stream>>>(dlt, xc, xz, D_param, yb);

  // 10. out_proj + residual: out[4096,1024] = x + yb[4096,2048] * woutb[1024,2048]^T
  gemm_bf16_wmma<1><<<dim3(D_MODEL / 128, NROWS / 128), 256, 0, stream>>>(
      yb, D_INNER, woutb, D_INNER, out, D_MODEL, x, D_MODEL,
      NROWS, D_MODEL, D_INNER);
}
